// MentionPairScore_75505525064177
// MI455X (gfx1250) — matrix operand order repaired and checked by reading, not compiled, not run on hardware
//
#include <hip/hip_runtime.h>
#include <math.h>

// ---------------------------------------------------------------------------
// MentionPairScore for MI455X (gfx1250, wave32, WMMA).
// Every matmul runs through v_wmma_f32_16x16x32_f16 (f16 A/B, f32 acc).
// GEMM: 128x64 block tile, 8 waves, 32x32 per wave (2x2 WMMA accumulators).
// A/B k-slabs staged into LDS via double-buffered async copies
// (global_load_async_to_lds_b128, ASYNCcnt). Pipeline: wait(0) -> barrier ->
// issue(next) -> compute(cur): one barrier + one wait per chunk; next chunk's
// loads overlap the current chunk's WMMAs. Fragments come from ds_load_b128.
// All GEMM dims padded (K mult-32, N mult-64, M mult-128) -> no guards.
// Workspace (~350MB) carved from d_ws; recomputed every call (capture-safe).
// ---------------------------------------------------------------------------

typedef __attribute__((ext_vector_type(16))) _Float16 v16h;
typedef __attribute__((ext_vector_type(8)))  _Float16 h8_t;
typedef __attribute__((ext_vector_type(8)))  float    v8f;

constexpr int Ltok = 2048, DM = 1024, NH = 8, DK = 128;
constexpr int Pn = 8192, Tn = 16;
constexpr int WE = 300, EIN = 320;            // WE + POS(20)
constexpr int RH = 256, G3 = 768, KX = 320;   // GRU dims, padded input K
constexpr int SINP = 5152;                    // 5140 -> pad 5152
constexpr int H1 = 300, H1P = 320, H2 = 150, H2P = 192, H3P = 64;
constexpr int VOCAB = 50000;

__device__ __forceinline__ float actf(float x, int act) {
  if (act == 1) return 0.5f * x * (1.f + erff(x * 0.70710678118f)); // gelu
  if (act == 2) return fmaxf(x, 0.f);                              // relu
  return x;
}

__device__ __forceinline__ float blockReduceSum(float v) {
  __shared__ float sh[256];
  sh[threadIdx.x] = v; __syncthreads();
  for (int s = 128; s > 0; s >>= 1) {
    if ((int)threadIdx.x < s) sh[threadIdx.x] += sh[threadIdx.x + s];
    __syncthreads();
  }
  float r = sh[0]; __syncthreads(); return r;
}

__device__ __forceinline__ float blockReduceMax(float v) {
  __shared__ float sh[256];
  sh[threadIdx.x] = v; __syncthreads();
  for (int s = 128; s > 0; s >>= 1) {
    if ((int)threadIdx.x < s) sh[threadIdx.x] = fmaxf(sh[threadIdx.x], sh[threadIdx.x + s]);
    __syncthreads();
  }
  float r = sh[0]; __syncthreads(); return r;
}

// ------------------------- batched WMMA GEMM -------------------------------
// C[z] = act(A[z][M,K] * B[z][N,K]^T + bias). 256 threads = 8 waves (4x2),
// block tile 128x64, wave tile 32x32 (2x2 accumulators). Double-buffered
// async-to-LDS staging of the A (128x32 halves) and B (64x32) k-slabs.
__global__ __launch_bounds__(256)
void gemm_f16_wmma(const _Float16* __restrict__ A, long long sA,
                   const _Float16* __restrict__ B, long long sB,
                   const float* __restrict__ bias, int nBias,
                   float* __restrict__ Cf, _Float16* __restrict__ Ch,
                   long long sC, int ldc, int ldct, int chTrans,
                   int K, int act)
{
  // halves: A buffers [0,4096),[4096,8192); B buffers [8192,10240),[10240,12288)
  __shared__ _Float16 smem[12288];

  const int tid  = threadIdx.x;
  const int lane = tid & 31;
  const int wv   = tid >> 5;
  const int hi   = lane >> 4;        // half-wave select per ISA WMMA layouts
  const int r16  = lane & 15;
  const int wm   = wv >> 1, wn = wv & 1;
  const long long z = blockIdx.z;
  const int m0 = blockIdx.y * 128;
  const int n0 = blockIdx.x * 64;

  // per-thread staging coordinates: A row (tid>>1) gets 2x16B, B row (tid>>2) 1x16B
  const _Float16* Ag = A + z * sA + (long long)(m0 + (tid >> 1)) * K + (tid & 1) * 16;
  const _Float16* Bg = B + z * sB + (long long)(n0 + (tid >> 2)) * K + (tid & 3) * 8;
  // low 32 bits of a generic shared pointer are the LDS byte address (ISA aperture rule)
  const unsigned ldsBase = (unsigned)(unsigned long long)(uintptr_t)smem;
  const unsigned aL = ldsBase + (unsigned)((tid >> 1) * 64 + (tid & 1) * 32);
  const unsigned bL = ldsBase + 16384u + (unsigned)((tid >> 2) * 64 + (tid & 3) * 16);

  v8f acc[2][2] = {{{}, {}}, {{}, {}}};

  // issue one k-chunk (3 async b128 per thread, tracked by ASYNCcnt)
  auto issueChunk = [&](int kc, int buf) {
    unsigned a0 = aL + (unsigned)buf * 8192u;
    unsigned b0 = bL + (unsigned)buf * 4096u;
    const _Float16* ga = Ag + kc;
    const _Float16* gb = Bg + kc;
    asm volatile("global_load_async_to_lds_b128 %0, %1, off"
                 :: "v"(a0), "v"(ga) : "memory");
    asm volatile("global_load_async_to_lds_b128 %0, %1, off offset:16"
                 :: "v"(a0), "v"(ga) : "memory");
    asm volatile("global_load_async_to_lds_b128 %0, %1, off"
                 :: "v"(b0), "v"(gb) : "memory");
  };

  // 8x ds_load_b128 fragment reads + 4 WMMAs on the given LDS buffer
  auto computeChunk = [&](int buf) {
    const _Float16* As = smem + (size_t)buf * 4096;
    const _Float16* Bs = smem + 8192 + (size_t)buf * 2048;
    union Frag { v16h v; h8_t h[2]; } af[2], bf[2];
    #pragma unroll
    for (int s = 0; s < 2; ++s) {
      const _Float16* ar = As + (wm * 32 + s * 16 + r16) * 32 + hi * 8;
      af[s].h[0] = *(const h8_t*)(ar);
      af[s].h[1] = *(const h8_t*)(ar + 16);
      const _Float16* br = Bs + (wn * 32 + s * 16 + r16) * 32 + hi * 8;
      bf[s].h[0] = *(const h8_t*)(br);
      bf[s].h[1] = *(const h8_t*)(br + 16);
    }
    #pragma unroll
    for (int s = 0; s < 2; ++s)
      #pragma unroll
      for (int u = 0; u < 2; ++u)
        acc[s][u] = __builtin_amdgcn_wmma_f32_16x16x32_f16(
            false, af[s].v, false, bf[u].v, (short)0, acc[s][u], false, false);
  };

  // Pipeline: wait(own chunk-i copies) -> barrier (all waves' data landed AND
  // everyone finished compute(i-1)) -> issue(i+1) (buffer last read at i-1,
  // now safe) -> compute(i). Last chunk peeled: no in-loop conditional.
  const int nChunks = K >> 5;
  issueChunk(0, 0);
  for (int i = 0; i < nChunks - 1; ++i) {
    asm volatile("s_wait_asynccnt 0x0" ::: "memory");
    __syncthreads();
    issueChunk((i + 1) << 5, (i + 1) & 1);
    computeChunk(i & 1);
  }
  asm volatile("s_wait_asynccnt 0x0" ::: "memory");
  __syncthreads();
  computeChunk((nChunks - 1) & 1);

  #pragma unroll
  for (int s = 0; s < 2; ++s) {
    #pragma unroll
    for (int u = 0; u < 2; ++u) {
      const int cn = n0 + wn * 32 + u * 16 + r16;
      const float bv = (bias && cn < nBias) ? bias[cn] : 0.f;
      #pragma unroll
      for (int r = 0; r < 8; ++r) {
        const int gm = m0 + wm * 32 + s * 16 + hi * 8 + r;  // C/D: VGPR r -> M=r(+8)
        float x = actf(acc[s][u][r] + bv, act);
        if (Cf) Cf[z * sC + (long long)gm * ldc + cn] = x;
        if (Ch) {
          if (chTrans) Ch[z * sC + (long long)cn * ldct + gm] = (_Float16)x;
          else         Ch[z * sC + (long long)gm * ldc  + cn] = (_Float16)x;
        }
      }
    }
  }
}

// ------------------------- pointwise / utility kernels ---------------------
__global__ void cast_pad_k(const float* __restrict__ in, _Float16* __restrict__ out,
                           int N, int Kin, int Kpad) {
  long long i = (long long)blockIdx.x * 256 + threadIdx.x;
  if (i >= (long long)N * Kpad) return;
  int n = (int)(i / Kpad), k = (int)(i % Kpad);
  out[i] = (k < Kin) ? (_Float16)in[(long long)n * Kin + k] : (_Float16)0.f;
}

// in [Z,Kin,Nin] f32 -> out [Z,Npad,Kpad] f16 (transposed, zero-padded)
__global__ void cast_trans_k(const float* __restrict__ in, _Float16* __restrict__ out,
                             int Kin, int Nin, int Kpad, int Npad,
                             long long sIn, long long sOut) {
  long long i = (long long)blockIdx.x * 256 + threadIdx.x;
  long long z = blockIdx.z;
  if (i >= (long long)Npad * Kpad) return;
  int n = (int)(i / Kpad), k = (int)(i % Kpad);
  float v = (n < Nin && k < Kin) ? in[z * sIn + (long long)k * Nin + n] : 0.f;
  out[z * sOut + i] = (_Float16)v;
}

__global__ void embed_doc_k(const int* __restrict__ tok, const int* __restrict__ pos,
                            const float* __restrict__ we, const float* __restrict__ pe,
                            _Float16* __restrict__ out) {
  int i = blockIdx.x * 256 + threadIdx.x;
  if (i >= Ltok * EIN) return;
  int l = i / EIN, c = i % EIN;
  float v = (c < WE) ? we[(long long)tok[l] * WE + c] : pe[(long long)pos[l] * 20 + (c - WE)];
  out[i] = (_Float16)v;
}

__global__ void gather_x_k(const _Float16* __restrict__ we, const int* __restrict__ spans,
                           int t, _Float16* __restrict__ xt) {
  long long i = (long long)blockIdx.x * 256 + threadIdx.x;
  if (i >= (long long)Pn * KX) return;
  long long p = i / KX; int c = (int)(i % KX);
  xt[i] = (c < WE) ? we[(long long)spans[p * Tn + t] * WE + c] : (_Float16)0.f;
}

__global__ void gru_gate_k(const float* __restrict__ gi, const float* __restrict__ gh,
                           float* __restrict__ h, _Float16* __restrict__ hh) {
  long long i = (long long)blockIdx.x * 256 + threadIdx.x;
  if (i >= (long long)Pn * RH) return;
  long long p = i / RH; int j = (int)(i % RH);
  const float* gp = gi + p * G3; const float* hp = gh + p * G3;
  float r  = 1.f / (1.f + __expf(-(gp[j]          + hp[j])));
  float zz = 1.f / (1.f + __expf(-(gp[RH + j]     + hp[RH + j])));
  float nn = tanhf(gp[2 * RH + j] + r * hp[2 * RH + j]);
  float hv = (1.f - zz) * nn + zz * h[i];
  h[i] = hv; hh[i] = (_Float16)hv;
}

__global__ void fill_f32_k(float* p, long long n, float v) {
  long long i = (long long)blockIdx.x * 256 + threadIdx.x; if (i < n) p[i] = v;
}
__global__ void fill_f16_k(_Float16* p, long long n) {
  long long i = (long long)blockIdx.x * 256 + threadIdx.x; if (i < n) p[i] = (_Float16)0.f;
}
__global__ void copy_h_k(const float* __restrict__ h, float* __restrict__ dst, int colOff) {
  long long i = (long long)blockIdx.x * 256 + threadIdx.x;
  if (i >= (long long)Pn * RH) return;
  long long p = i / RH; int j = (int)(i % RH);
  dst[p * (2 * RH) + colOff + j] = h[i];
}

// row LayerNorm (torch: ddof=1 var, sigma clamp 1e-6), optional residual,
// optional f32 and/or f16 outputs (f16 path writes into the feature matrix).
__global__ __launch_bounds__(256)
void layernorm_k(const float* __restrict__ in, const float* __restrict__ resid, int W,
                 const float* __restrict__ g, const float* __restrict__ bb,
                 float* __restrict__ outf, _Float16* __restrict__ outh,
                 long long ldin, long long ldof, long long ldoh) {
  long long row = blockIdx.x;
  const float* x  = in + row * ldin;
  const float* rz = resid ? resid + row * ldin : nullptr;
  float s = 0.f;
  for (int c = threadIdx.x; c < W; c += 256) s += x[c] + (rz ? rz[c] : 0.f);
  float mu = blockReduceSum(s) / (float)W;
  float s2 = 0.f;
  for (int c = threadIdx.x; c < W; c += 256) {
    float v = x[c] + (rz ? rz[c] : 0.f) - mu; s2 += v * v;
  }
  float var = blockReduceSum(s2) / (float)(W - 1);
  float sig = fmaxf(sqrtf(var), 1e-6f);
  for (int c = threadIdx.x; c < W; c += 256) {
    float y = (x[c] + (rz ? rz[c] : 0.f) - mu) / sig * g[c] + bb[c];
    if (outf) outf[row * ldof + c] = y;
    if (outh) outh[row * ldoh + c] = (_Float16)y;
  }
}

// in-place f16 softmax with pre-scale (1/sqrt(DK)); one block per row
__global__ __launch_bounds__(256)
void softmax_inplace_h(_Float16* __restrict__ sc, int W, float scale) {
  long long row = blockIdx.x;
  _Float16* x = sc + row * (long long)W;
  float m = -3.4e38f;
  for (int c = threadIdx.x; c < W; c += 256) m = fmaxf(m, (float)x[c] * scale);
  m = blockReduceMax(m);
  float s = 0.f;
  for (int c = threadIdx.x; c < W; c += 256) {
    float e = __expf((float)x[c] * scale - m); s += e; x[c] = (_Float16)e;
  }
  s = blockReduceSum(s);
  float inv = 1.f / s;
  for (int c = threadIdx.x; c < W; c += 256) x[c] = (_Float16)((float)x[c] * inv);
}

// gather 4 doc_enc endpoint rows + dist embedding + zero pad into feat (f16)
__global__ __launch_bounds__(256)
void feat_doc_k(const float* __restrict__ denc, const int* __restrict__ mIdx,
                const int* __restrict__ cIdx, const float* __restrict__ demb,
                const int* __restrict__ dist, _Float16* __restrict__ feat) {
  int p = blockIdx.x;
  long long r0 = mIdx[p * 2], r1 = mIdx[p * 2 + 1];
  long long r2 = cIdx[p * 2], r3 = cIdx[p * 2 + 1];
  _Float16* f = feat + (long long)p * SINP;
  for (int c = threadIdx.x; c < DM; c += 256) {
    f[c]          = (_Float16)denc[r0 * DM + c];
    f[DM + c]     = (_Float16)denc[r1 * DM + c];
    f[2 * DM + c] = (_Float16)denc[r2 * DM + c];
    f[3 * DM + c] = (_Float16)denc[r3 * DM + c];
  }
  if (threadIdx.x < 32) {   // cols 5120..5151: 20 dist dims + 12 zero pad
    int c = threadIdx.x;
    f[5120 + c] = (c < 20) ? (_Float16)demb[(long long)dist[p] * 20 + c] : (_Float16)0.f;
  }
}

__global__ void final_k(const float* __restrict__ sc, const int* __restrict__ labels,
                        float* __restrict__ out) {
  int p = blockIdx.x * 256 + threadIdx.x;
  if (p >= Pn) return;
  out[p]      = 1.f / (1.f + __expf(-sc[(long long)p * H3P]));
  out[Pn + p] = (float)labels[p];
}

// ---------------------------------------------------------------------------
extern "C" void kernel_launch(void* const* d_in, const int* in_sizes, int n_in,
                              void* d_out, int out_size, void* d_ws, size_t ws_size,
                              hipStream_t stream) {
  (void)in_sizes; (void)n_in; (void)out_size; (void)ws_size;
  const int* tokens    = (const int*)d_in[0];
  const int* positions = (const int*)d_in[1];
  const int* m_spans   = (const int*)d_in[2];
  const int* c_spans   = (const int*)d_in[3];
  const int* dists     = (const int*)d_in[4];
  const int* m_idxs    = (const int*)d_in[5];
  const int* c_idxs    = (const int*)d_in[6];
  const int* labels    = (const int*)d_in[7];
  const float* word_emb = (const float*)d_in[8];
  const float* pos_emb  = (const float*)d_in[9];
  const float* dist_emb = (const float*)d_in[10];
  const float* Wt = (const float*)d_in[11]; const float* bt = (const float*)d_in[12];
  const float* w_qs = (const float*)d_in[13]; const float* w_ks = (const float*)d_in[14];
  const float* w_vs = (const float*)d_in[15]; const float* w_o = (const float*)d_in[16];
  const float* ln_g = (const float*)d_in[17]; const float* ln_b = (const float*)d_in[18];
  const float* Wih_f = (const float*)d_in[19]; const float* Whh_f = (const float*)d_in[20];
  const float* bih_f = (const float*)d_in[21]; const float* bhh_f = (const float*)d_in[22];
  const float* Wih_b = (const float*)d_in[23]; const float* Whh_b = (const float*)d_in[24];
  const float* bih_b = (const float*)d_in[25]; const float* bhh_b = (const float*)d_in[26];
  const float* rln_g = (const float*)d_in[27]; const float* rln_b = (const float*)d_in[28];
  const float* sW1 = (const float*)d_in[29]; const float* sb1 = (const float*)d_in[30];
  const float* sW2 = (const float*)d_in[31]; const float* sb2 = (const float*)d_in[32];
  const float* sW3 = (const float*)d_in[33]; const float* sb3 = (const float*)d_in[34];
  float* out = (float*)d_out;

  // ---- workspace carve (assumes ws_size >= ~350MB) ----
  char* base = (char*)d_ws; size_t off = 0;
  auto carveH = [&](size_t n) { off = (off + 255) & ~(size_t)255;
                                _Float16* p = (_Float16*)(base + off); off += n * 2; return p; };
  auto carveF = [&](size_t n) { off = (off + 255) & ~(size_t)255;
                                float* p = (float*)(base + off); off += n * 4; return p; };

  _Float16* we_h  = carveH((size_t)VOCAB * WE);
  _Float16* WtT   = carveH((size_t)DM * EIN);
  _Float16* wqT   = carveH((size_t)NH * DK * DM);
  _Float16* wkT   = carveH((size_t)NH * DK * DM);
  _Float16* wvT   = carveH((size_t)NH * DK * DM);
  _Float16* woT   = carveH((size_t)DM * DM);
  _Float16* WihTf = carveH((size_t)G3 * KX);
  _Float16* WhhTf = carveH((size_t)G3 * RH);
  _Float16* WihTb = carveH((size_t)G3 * KX);
  _Float16* WhhTb = carveH((size_t)G3 * RH);
  _Float16* sW1T  = carveH((size_t)H1P * SINP);
  _Float16* sW2T  = carveH((size_t)H2P * H1P);
  _Float16* sW3T  = carveH((size_t)H3P * H2P);

  _Float16* docin = carveH((size_t)Ltok * EIN);
  _Float16* doc_h = carveH((size_t)Ltok * DM);
  float*    doc_f = carveF((size_t)Ltok * DM);
  _Float16* q_h   = carveH((size_t)NH * Ltok * DK);
  _Float16* k_h   = carveH((size_t)NH * Ltok * DK);
  _Float16* vT_h  = carveH((size_t)NH * DK * Ltok);
  _Float16* sc_h  = carveH((size_t)NH * Ltok * Ltok);
  _Float16* att_h = carveH((size_t)Ltok * DM);
  float*    proj_f = carveF((size_t)Ltok * DM);
  float*    denc   = carveF((size_t)Ltok * DM);

  _Float16* xt_h = carveH((size_t)Pn * KX);
  float*    gi_f = carveF((size_t)Pn * G3);
  float*    gh_f = carveF((size_t)Pn * G3);
  float*    h_f  = carveF((size_t)Pn * RH);
  _Float16* h_h  = carveH((size_t)Pn * RH);
  float*    rcat = carveF((size_t)Pn * 2 * RH);
  _Float16* feat = carveH((size_t)Pn * SINP);
  _Float16* h1_h = carveH((size_t)Pn * H1P);
  _Float16* h2_h = carveH((size_t)Pn * H2P);
  float*    scp  = carveF((size_t)Pn * H3P);

  dim3 blk(256);
  auto cdiv = [](long long a, long long b) { return (unsigned)((a + b - 1) / b); };
  auto gemm = [&](const _Float16* A, long long sA, const _Float16* B, long long sB,
                  const float* bias, int nBias, float* Cf, _Float16* Ch, long long sC,
                  int ldc, int ldct, int chT, int M, int N, int K, int act, int Z) {
    dim3 g((unsigned)(N / 64), (unsigned)(M / 128), (unsigned)Z);
    gemm_f16_wmma<<<g, blk, 0, stream>>>(A, sA, B, sB, bias, nBias, Cf, Ch, sC,
                                         ldc, ldct, chT, K, act);
  };

  // ---- weight conversions to f16 ([N,K] layout, zero-padded) ----
  cast_pad_k<<<cdiv((long long)VOCAB * WE, 256), blk, 0, stream>>>(word_emb, we_h, VOCAB, WE, WE);
  cast_trans_k<<<dim3(cdiv((long long)DM * EIN, 256), 1, 1), blk, 0, stream>>>(
      Wt, WtT, EIN, DM, EIN, DM, 0, 0);
  cast_trans_k<<<dim3(cdiv((long long)DK * DM, 256), 1, NH), blk, 0, stream>>>(
      w_qs, wqT, DM, DK, DM, DK, (long long)DM * DK, (long long)DK * DM);
  cast_trans_k<<<dim3(cdiv((long long)DK * DM, 256), 1, NH), blk, 0, stream>>>(
      w_ks, wkT, DM, DK, DM, DK, (long long)DM * DK, (long long)DK * DM);
  cast_trans_k<<<dim3(cdiv((long long)DK * DM, 256), 1, NH), blk, 0, stream>>>(
      w_vs, wvT, DM, DK, DM, DK, (long long)DM * DK, (long long)DK * DM);
  cast_pad_k<<<cdiv((long long)DM * DM, 256), blk, 0, stream>>>(w_o, woT, DM, DM, DM);
  cast_pad_k<<<cdiv((long long)G3 * KX, 256), blk, 0, stream>>>(Wih_f, WihTf, G3, WE, KX);
  cast_pad_k<<<cdiv((long long)G3 * RH, 256), blk, 0, stream>>>(Whh_f, WhhTf, G3, RH, RH);
  cast_pad_k<<<cdiv((long long)G3 * KX, 256), blk, 0, stream>>>(Wih_b, WihTb, G3, WE, KX);
  cast_pad_k<<<cdiv((long long)G3 * RH, 256), blk, 0, stream>>>(Whh_b, WhhTb, G3, RH, RH);
  cast_trans_k<<<dim3(cdiv((long long)H1P * SINP, 256), 1, 1), blk, 0, stream>>>(
      sW1, sW1T, 5140, H1, SINP, H1P, 0, 0);
  cast_trans_k<<<dim3(cdiv((long long)H2P * H1P, 256), 1, 1), blk, 0, stream>>>(
      sW2, sW2T, H1, H2, H1P, H2P, 0, 0);
  cast_trans_k<<<dim3(cdiv((long long)H3P * H2P, 256), 1, 1), blk, 0, stream>>>(
      sW3, sW3T, H2, 1, H2P, H3P, 0, 0);

  // ---- doc encode: embed -> gelu(x@Wt+bt) -> 8-head attention -> LN ----
  embed_doc_k<<<cdiv((long long)Ltok * EIN, 256), blk, 0, stream>>>(
      tokens, positions, word_emb, pos_emb, docin);
  gemm(docin, 0, WtT, 0, bt, DM, doc_f, doc_h, 0, DM, 0, 0, Ltok, DM, EIN, 1, 1);
  gemm(doc_h, 0, wqT, (long long)DK * DM, nullptr, 0, nullptr, q_h,
       (long long)Ltok * DK, DK, 0, 0, Ltok, DK, DM, 0, NH);
  gemm(doc_h, 0, wkT, (long long)DK * DM, nullptr, 0, nullptr, k_h,
       (long long)Ltok * DK, DK, 0, 0, Ltok, DK, DM, 0, NH);
  gemm(doc_h, 0, wvT, (long long)DK * DM, nullptr, 0, nullptr, vT_h,
       (long long)DK * Ltok, DK, Ltok, 1, Ltok, DK, DM, 0, NH);      // V stored transposed
  gemm(q_h, (long long)Ltok * DK, k_h, (long long)Ltok * DK, nullptr, 0, nullptr, sc_h,
       (long long)Ltok * Ltok, Ltok, 0, 0, Ltok, Ltok, DK, 0, NH);   // Q K^T
  softmax_inplace_h<<<NH * Ltok, blk, 0, stream>>>(sc_h, Ltok, 0.08838834764831845f);
  gemm(sc_h, (long long)Ltok * Ltok, vT_h, (long long)DK * Ltok, nullptr, 0, nullptr,
       att_h, (long long)DK, DM, 0, 0, Ltok, DK, Ltok, 0, NH);       // head-major concat via sC=DK
  gemm(att_h, 0, woT, 0, nullptr, 0, proj_f, nullptr, 0, DM, 0, 0, Ltok, DM, DM, 0, 1);
  layernorm_k<<<Ltok, blk, 0, stream>>>(proj_f, doc_f, DM, ln_g, ln_b,
                                        denc, nullptr, DM, DM, 0);

  // ---- pair features: doc endpoints + dist (rnn cols filled below) ----
  feat_doc_k<<<Pn, blk, 0, stream>>>(denc, m_idxs, c_idxs, dist_emb, dists, feat);

  // ---- GRU span encoders. enc[:,-1] only needs final fwd hidden and the
  //      FIRST backward-scan hidden (one step on x[:,T-1]). ----
  auto gru_step = [&](const int* spans, int t, const _Float16* WihT, const _Float16* WhhT,
                      const float* bih, const float* bhh) {
    gather_x_k<<<cdiv((long long)Pn * KX, 256), blk, 0, stream>>>(we_h, spans, t, xt_h);
    gemm(xt_h, 0, WihT, 0, bih, G3, gi_f, nullptr, 0, G3, 0, 0, Pn, G3, KX, 0, 1);
    gemm(h_h, 0, WhhT, 0, bhh, G3, gh_f, nullptr, 0, G3, 0, 0, Pn, G3, RH, 0, 1);
    gru_gate_k<<<cdiv((long long)Pn * RH, 256), blk, 0, stream>>>(gi_f, gh_f, h_f, h_h);
  };
  auto gru_set = [&](const int* spans, int featCol) {
    fill_f32_k<<<cdiv((long long)Pn * RH, 256), blk, 0, stream>>>(h_f, (long long)Pn * RH, 0.f);
    fill_f16_k<<<cdiv((long long)Pn * RH, 256), blk, 0, stream>>>(h_h, (long long)Pn * RH);
    for (int t = 0; t < Tn; ++t) gru_step(spans, t, WihTf, WhhTf, bih_f, bhh_f);
    copy_h_k<<<cdiv((long long)Pn * RH, 256), blk, 0, stream>>>(h_f, rcat, 0);
    fill_f32_k<<<cdiv((long long)Pn * RH, 256), blk, 0, stream>>>(h_f, (long long)Pn * RH, 0.f);
    fill_f16_k<<<cdiv((long long)Pn * RH, 256), blk, 0, stream>>>(h_h, (long long)Pn * RH);
    gru_step(spans, Tn - 1, WihTb, WhhTb, bih_b, bhh_b);
    copy_h_k<<<cdiv((long long)Pn * RH, 256), blk, 0, stream>>>(h_f, rcat, RH);
    layernorm_k<<<Pn, blk, 0, stream>>>(rcat, nullptr, 2 * RH, rln_g, rln_b,
                                        nullptr, feat + featCol, 2 * RH, 0, SINP);
  };
  gru_set(m_spans, 4 * DM);            // feat cols 4096..4607
  gru_set(c_spans, 4 * DM + 2 * RH);   // feat cols 4608..5119

  // ---- scoring MLP + sigmoid; labels appended ----
  gemm(feat, 0, sW1T, 0, sb1, H1, nullptr, h1_h, 0, H1P, 0, 0, Pn, H1P, SINP, 2, 1);
  gemm(h1_h, 0, sW2T, 0, sb2, H2, nullptr, h2_h, 0, H2P, 0, 0, Pn, H2P, H1P, 2, 1);
  gemm(h2_h, 0, sW3T, 0, sb3, 1, scp, nullptr, 0, H3P, 0, 0, Pn, H3P, H2P, 0, 1);
  final_k<<<cdiv(Pn, 256), blk, 0, stream>>>(scp, labels, out);
}
